// RNN_80977313399519
// MI455X (gfx1250) — matrix operand order repaired and checked by reading, not compiled, not run on hardware
//
#include <hip/hip_runtime.h>
#include <hip/hip_bf16.h>

typedef _Float16 v16h __attribute__((ext_vector_type(16)));
typedef _Float16 h8   __attribute__((ext_vector_type(8)));
typedef float    v8f  __attribute__((ext_vector_type(8)));
typedef float    f4   __attribute__((ext_vector_type(4)));
typedef unsigned int u32x4 __attribute__((ext_vector_type(4)));
typedef int          i32x8 __attribute__((ext_vector_type(8)));
typedef int          i32x4 __attribute__((ext_vector_type(4)));

#define Bsz 256
#define Tsz 1000
#define Isz 64
#define Hsz 128
#define G3  384
#define HSTRIDE 136      // padded f16 row stride for h in LDS (272B = 17*16B)

// TDM x-staging ring: slab = 16 rows x 384 f32, padded +4 dwords per 256 dwords
#define NBUF       6
#define SLAB_DW    (16 * G3 + (16 * G3 / 256) * 4)   // 6240 dwords
#define SLAB_BYTES (SLAB_DW * 4)                      // 24960 bytes

#if defined(__HIP_DEVICE_COMPILE__) && \
    __has_builtin(__builtin_amdgcn_tensor_load_to_lds) && \
    __has_builtin(__builtin_amdgcn_s_wait_tensorcnt)
#define GRU_TDM 1
#else
#define GRU_TDM 0
#endif

static __device__ __forceinline__ v8f wmma_f16(v16h a, v16h b, v8f c) {
  // D = A(16x32 f16) * B(32x16 f16) + C(16x16 f32)
  return __builtin_amdgcn_wmma_f32_16x16x32_f16(false, a, false, b, (short)0, c,
                                                false, false);
}

static __device__ __forceinline__ float sigm(float x) {
  return __builtin_amdgcn_rcpf(1.0f + __expf(-x));
}
static __device__ __forceinline__ float tanh_fast(float x) {
  const float e2 = __expf(-2.0f * fabsf(x));
  return copysignf((1.0f - e2) * __builtin_amdgcn_rcpf(1.0f + e2), x);
}

// A-operand pack: element e holds K = kb + (e<8 ? e : e+8), from two groups of 8
static __device__ __forceinline__ v16h packA(f4 p0, f4 p1, f4 p2, f4 p3) {
  v16h av;
  #pragma unroll
  for (int e = 0; e < 4; ++e) {
    av[e]      = (_Float16)p0[e];
    av[4 + e]  = (_Float16)p1[e];
    av[8 + e]  = (_Float16)p2[e];
    av[12 + e] = (_Float16)p3[e];
  }
  return av;
}

#if GRU_TDM
// Issue one TDM 2D-tile load: 16 rows x 1536B, row stride T*3H floats,
// LDS padding 4 dwords per 256 dwords. D# layout per CDNA5 ISA ch.8.
static __device__ __forceinline__ void tdm_issue(const float* gsrc,
                                                 unsigned lds_byte_off) {
  const unsigned long long ga = (unsigned long long)(uintptr_t)gsrc;
  u32x4 g0;
  g0[0] = 1u;                                   // count=1, user descriptor
  g0[1] = lds_byte_off;                         // lds_addr
  g0[2] = (unsigned)ga;                         // global_addr[31:0]
  g0[3] = (unsigned)(ga >> 32) | 0x80000000u;   // global_addr[56:32] | type=2
  i32x8 g1;
  g1[0] = (2 << 16) | (1 << 20) | (7 << 22) | (3 << 25);
      // data_size=4B | pad_enable | pad_interval=256dw | pad_amount=4dw
  g1[1] = (int)(384u << 16);                    // tensor_dim0 = 384 (lo16)
  g1[2] = (int)(16u << 16);                     // dim0 hi=0 | tensor_dim1=16 lo
  g1[3] = (int)(384u << 16);                    // dim1 hi=0 | tile_dim0=384
  g1[4] = 16;                                   // tile_dim1=16, tile_dim2=0
  g1[5] = Tsz * G3;                             // tensor_dim0_stride lo32
  g1[6] = 0;                                    // stride hi | dim1_stride lo
  g1[7] = 0;
  const i32x4 z4 = {0, 0, 0, 0};                // groups 2/3 unused (2D tile)
  const i32x8 z8 = {0, 0, 0, 0, 0, 0, 0, 0};    // extra group (6-arg form)
  __builtin_amdgcn_tensor_load_to_lds(g0, g1, z4, z4, z8, 0);
}
#endif

// ---------------------------------------------------------------------------
// Phase 1: x_proj[(b*T+t), 0:384] = input[b,t,:] @ W_ih^T + b_ih
// Transposed orientation: D^T = W_ih(A, static) x input^T(B).
// ---------------------------------------------------------------------------
__global__ __launch_bounds__(256) void gru_xproj_kernel(
    const float* __restrict__ x, const float* __restrict__ Wih,
    const float* __restrict__ bih, float* __restrict__ xproj)
{
  const int lane  = threadIdx.x & 31;
  const int wave  = threadIdx.x >> 5;
  const int nl    = lane & 15;
  const int hi    = lane >> 4;
  const int nbase = wave * 48;
  const int cb    = nbase + 8 * hi;

  v16h Aop[3][2];
  v8f bias8[3];
  #pragma unroll
  for (int j = 0; j < 3; ++j) {
    const float* wr = Wih + (nbase + 16 * j + nl) * Isz;
    #pragma unroll
    for (int c = 0; c < 2; ++c) {
      const int kb = 32 * c + 8 * hi;
      Aop[j][c] = packA(*(const f4*)(wr + kb), *(const f4*)(wr + kb + 4),
                        *(const f4*)(wr + kb + 16), *(const f4*)(wr + kb + 20));
    }
    const f4 b0 = *(const f4*)&bih[cb + 16 * j];
    const f4 b1 = *(const f4*)&bih[cb + 16 * j + 4];
    #pragma unroll
    for (int r = 0; r < 4; ++r) { bias8[j][r] = b0[r]; bias8[j][4 + r] = b1[r]; }
  }

  const int ntiles = (Bsz * Tsz) / 16;  // 16000
  for (int tile = blockIdx.x; tile < ntiles; tile += gridDim.x) {
    const float* xr = x + (size_t)(tile * 16 + nl) * Isz;
    v16h Bop[2];
    #pragma unroll
    for (int c = 0; c < 2; ++c) {
      const int kb = 32 * c + 8 * hi;
      f4 q0 = *(const f4*)(xr + kb);
      f4 q1 = *(const f4*)(xr + kb + 4);
      f4 q2 = *(const f4*)(xr + kb + 16);
      f4 q3 = *(const f4*)(xr + kb + 20);
      v16h bv;
      #pragma unroll
      for (int e = 0; e < 4; ++e) {
        bv[e]      = (_Float16)q0[e];
        bv[4 + e]  = (_Float16)q1[e];
        bv[8 + e]  = (_Float16)q2[e];
        bv[12 + e] = (_Float16)q3[e];
      }
      Bop[c] = bv;
    }
    float* orow = xproj + (size_t)(tile * 16 + nl) * G3 + cb;
    #pragma unroll
    for (int j = 0; j < 3; ++j) {
      v8f acc = wmma_f16(Aop[j][0], Bop[0], bias8[j]);  // C = bias, free init
      acc = wmma_f16(Aop[j][1], Bop[1], acc);
      f4 o0, o1;
      #pragma unroll
      for (int r = 0; r < 4; ++r) { o0[r] = acc[r]; o1[r] = acc[4 + r]; }
      *(f4*)(orow + 16 * j)     = o0;
      *(f4*)(orow + 16 * j + 4) = o1;
    }
  }
}

// ---------------------------------------------------------------------------
// Phase 2: GRU scan. 16 WGs x 16 batch rows, T=1000 sequential steps.
// hg^T = W_hh(A, static VGPRs) x h^T(B, LDS). Lane owns batch row lane%16
// and 8 contiguous H-cols. x_proj streamed via TDM ring (or reg pipeline).
// ---------------------------------------------------------------------------
__global__ __launch_bounds__(256) void gru_scan_kernel(
    const float* __restrict__ Whh, const float* __restrict__ bhh,
    const float* __restrict__ xproj, float* __restrict__ out)
{
  const int lane = threadIdx.x & 31;
  const int wave = threadIdx.x >> 5;
  const int nl   = lane & 15;
  const int hi   = lane >> 4;
  const int cb   = wave * 16 + 8 * hi;
  const int brow = blockIdx.x * 16 + nl;

  __shared__ _Float16 hbuf[16 * HSTRIDE];

  v16h Aop[3][4];
  v8f bias8[3];
  #pragma unroll
  for (int g = 0; g < 3; ++g) {
    const float* wr = Whh + (g * Hsz + wave * 16 + nl) * Hsz;
    #pragma unroll
    for (int c = 0; c < 4; ++c) {
      const int kb = 32 * c + 8 * hi;
      Aop[g][c] = packA(*(const f4*)(wr + kb), *(const f4*)(wr + kb + 4),
                        *(const f4*)(wr + kb + 16), *(const f4*)(wr + kb + 20));
    }
    const f4 b0 = *(const f4*)&bhh[g * Hsz + cb];
    const f4 b1 = *(const f4*)&bhh[g * Hsz + cb + 4];
    #pragma unroll
    for (int r = 0; r < 4; ++r) { bias8[g][r] = b0[r]; bias8[g][4 + r] = b1[r]; }
  }

  for (int i = threadIdx.x; i < 16 * HSTRIDE; i += 256)
    hbuf[i] = (_Float16)0.0f;
  float hprev[8];
  #pragma unroll
  for (int r = 0; r < 8; ++r) hprev[r] = 0.0f;

  float* sb = out + (size_t)brow * Tsz * Hsz + cb;
  float* hT = out + (size_t)Bsz * Tsz * Hsz + (size_t)brow * Hsz + cb;

#if GRU_TDM
  // ---------------- TDM-staged x pipeline (depth NBUF-1 slabs) --------------
  __shared__ float xstage[NBUF * SLAB_DW];

  // Per-lane read byte-offsets inside a slab (account for TDM padding).
  int xoff[3];
  #pragma unroll
  for (int g = 0; g < 3; ++g) {
    const int f = nl * G3 + g * Hsz + cb;
    xoff[g] = 4 * (f + 4 * (f >> 8));
  }

  const float* gx0 = xproj + (size_t)(blockIdx.x * 16) * Tsz * G3;
  const unsigned lds0 = (unsigned)(uintptr_t)xstage;
  if (wave == 0) {
    #pragma unroll
    for (int p = 0; p < NBUF - 1; ++p)
      tdm_issue(gx0 + (size_t)p * G3, lds0 + p * SLAB_BYTES);
    __builtin_amdgcn_s_wait_tensorcnt((short)(NBUF - 2));  // slab 0 resident
  }
  const float* gnext = gx0 + (size_t)(NBUF - 1) * G3;
  int slab = 0, islab = NBUF - 1;
  __syncthreads();

  for (int t = 0; t < Tsz; ++t) {
    // B operands: h^T from LDS; lane nb=nl, K = 32c+8hi+(0..7) and 32c+16+8hi+(0..7).
    v16h B[4];
    #pragma unroll
    for (int c = 0; c < 4; ++c) {
      const int base = nl * HSTRIDE + 32 * c + 8 * hi;
      const h8 lo = *(const h8*)&hbuf[base];
      const h8 hh = *(const h8*)&hbuf[base + 16];
      v16h bv;
      #pragma unroll
      for (int e = 0; e < 8; ++e) { bv[e] = lo[e]; bv[8 + e] = hh[e]; }
      B[c] = bv;
    }
    // x gate values from the resident slab.
    const char* sl = (const char*)xstage + slab * SLAB_BYTES;
    f4 xg0[3], xg1[3];
    #pragma unroll
    for (int g = 0; g < 3; ++g) {
      xg0[g] = *(const f4*)(sl + xoff[g]);
      xg1[g] = *(const f4*)(sl + xoff[g] + 16);
    }
    __syncthreads();  // hbuf reads done

    v8f a0 = wmma_f16(Aop[0][0], B[0], bias8[0]);
    v8f a1 = wmma_f16(Aop[1][0], B[0], bias8[1]);
    v8f a2 = wmma_f16(Aop[2][0], B[0], bias8[2]);
    #pragma unroll
    for (int c = 1; c < 4; ++c) {
      a0 = wmma_f16(Aop[0][c], B[c], a0);
      a1 = wmma_f16(Aop[1][c], B[c], a1);
      a2 = wmma_f16(Aop[2][c], B[c], a2);
    }

    f4 o0, o1;
    h8 hv;
    #pragma unroll
    for (int r = 0; r < 8; ++r) {
      const float xr_ = (r < 4) ? xg0[0][r & 3] : xg1[0][r & 3];
      const float xz_ = (r < 4) ? xg0[1][r & 3] : xg1[1][r & 3];
      const float xn_ = (r < 4) ? xg0[2][r & 3] : xg1[2][r & 3];
      const float rg = sigm(xr_ + a0[r]);
      const float zg = sigm(xz_ + a1[r]);
      const float ng = tanh_fast(xn_ + rg * a2[r]);
      const float hn = ng + zg * (hprev[r] - ng);
      hprev[r] = hn;
      hv[r] = (_Float16)hn;
      if (r < 4) o0[r & 3] = hn; else o1[r & 3] = hn;
    }
    *(h8*)&hbuf[nl * HSTRIDE + cb] = hv;
    *(f4*)sb = o0;
    *(f4*)(sb + 4) = o1;
    if (t == Tsz - 1) { *(f4*)hT = o0; *(f4*)(hT + 4) = o1; }

    if (wave == 0) {
      tdm_issue(gnext, lds0 + islab * SLAB_BYTES);           // slab t+NBUF-1
      __builtin_amdgcn_s_wait_tensorcnt((short)(NBUF - 2));  // slab t+1 done
    }
    if (t + NBUF < Tsz) gnext += G3;
    islab = (islab == NBUF - 1) ? 0 : islab + 1;
    slab  = (slab  == NBUF - 1) ? 0 : slab  + 1;
    sb += Hsz;
    __syncthreads();  // h + slab t+1 visible to all waves
  }
#else
  // ---------------- Fallback: register double-buffer + prefetch -------------
  const float* xb = xproj + (size_t)brow * Tsz * G3 + cb;

  f4 xc0[3][2], xc1[3][2];
  #pragma unroll
  for (int g = 0; g < 3; ++g) {
    xc0[g][0] = *(const f4*)(xb + g * Hsz);
    xc0[g][1] = *(const f4*)(xb + g * Hsz + 4);
  }
  __syncthreads();

  auto step = [&](int t, f4 (&xc)[3][2], f4 (&xn)[3][2]) {
    if (t + 6 < Tsz) {
      #pragma unroll
      for (int g = 0; g < 3; ++g)
        __builtin_prefetch(xb + 6 * G3 + g * Hsz, 0, 0);
    }
    if (t + 1 < Tsz) {
      #pragma unroll
      for (int g = 0; g < 3; ++g) {
        xn[g][0] = *(const f4*)(xb + G3 + g * Hsz);
        xn[g][1] = *(const f4*)(xb + G3 + g * Hsz + 4);
      }
    }
    v16h B[4];
    #pragma unroll
    for (int c = 0; c < 4; ++c) {
      const int base = nl * HSTRIDE + 32 * c + 8 * hi;
      const h8 lo = *(const h8*)&hbuf[base];
      const h8 hh = *(const h8*)&hbuf[base + 16];
      v16h bv;
      #pragma unroll
      for (int e = 0; e < 8; ++e) { bv[e] = lo[e]; bv[8 + e] = hh[e]; }
      B[c] = bv;
    }
    __syncthreads();

    v8f a0 = wmma_f16(Aop[0][0], B[0], bias8[0]);
    v8f a1 = wmma_f16(Aop[1][0], B[0], bias8[1]);
    v8f a2 = wmma_f16(Aop[2][0], B[0], bias8[2]);
    #pragma unroll
    for (int c = 1; c < 4; ++c) {
      a0 = wmma_f16(Aop[0][c], B[c], a0);
      a1 = wmma_f16(Aop[1][c], B[c], a1);
      a2 = wmma_f16(Aop[2][c], B[c], a2);
    }

    f4 o0, o1;
    h8 hv;
    #pragma unroll
    for (int r = 0; r < 8; ++r) {
      const float xr_ = (r < 4) ? xc[0][0][r & 3] : xc[0][1][r & 3];
      const float xz_ = (r < 4) ? xc[1][0][r & 3] : xc[1][1][r & 3];
      const float xn_ = (r < 4) ? xc[2][0][r & 3] : xc[2][1][r & 3];
      const float rg = sigm(xr_ + a0[r]);
      const float zg = sigm(xz_ + a1[r]);
      const float ng = tanh_fast(xn_ + rg * a2[r]);
      const float hn = ng + zg * (hprev[r] - ng);
      hprev[r] = hn;
      hv[r] = (_Float16)hn;
      if (r < 4) o0[r & 3] = hn; else o1[r & 3] = hn;
    }
    *(h8*)&hbuf[nl * HSTRIDE + cb] = hv;
    *(f4*)sb = o0;
    *(f4*)(sb + 4) = o1;
    if (t == Tsz - 1) { *(f4*)hT = o0; *(f4*)(hT + 4) = o1; }
    __syncthreads();

    xb += G3;
    sb += Hsz;
  };

  for (int t = 0; t < Tsz; t += 2) {
    step(t, xc0, xc1);
    step(t + 1, xc1, xc0);
  }
#endif
}

// ---------------------------------------------------------------------------
extern "C" void kernel_launch(void* const* d_in, const int* in_sizes, int n_in,
                              void* d_out, int out_size, void* d_ws,
                              size_t ws_size, hipStream_t stream) {
  (void)in_sizes; (void)n_in; (void)out_size; (void)ws_size;
  const float* input = (const float*)d_in[0];
  const float* W_ih  = (const float*)d_in[1];
  const float* W_hh  = (const float*)d_in[2];
  const float* b_ih  = (const float*)d_in[3];
  const float* b_hh  = (const float*)d_in[4];
  float* out   = (float*)d_out;
  float* xproj = (float*)d_ws;  // needs B*T*3H*4 = 393,216,000 bytes

  gru_xproj_kernel<<<512, 256, 0, stream>>>(input, W_ih, b_ih, xproj);
  gru_scan_kernel<<<Bsz / 16, 256, 0, stream>>>(W_hh, b_hh, xproj, out);
}
